// PositionAttentionModule_38989713113556
// MI455X (gfx1250) — matrix operand order repaired and checked by reading, not compiled, not run on hardware
//
#include <hip/hip_runtime.h>
#include <hip/hip_bf16.h>

// ---------------------------------------------------------------------------
// PositionAttentionModule for MI455X (gfx1250, wave32, WMMA f16->f32)
//   B=4, H=W=64, C=512, HW=4096, Cr=64
// ---------------------------------------------------------------------------

typedef __attribute__((ext_vector_type(16))) _Float16 v16h;
typedef __attribute__((ext_vector_type(8)))  _Float16 v8h;
typedef __attribute__((ext_vector_type(8)))  float    v8f;
typedef __attribute__((ext_vector_type(4)))  float    v4f;

#define HWD 4096
#define CD  512
#define CR  64
#define NB  4

// ---------------- WMMA wrapper -------------------------------------------
__device__ __forceinline__ v8f wmma16(v16h a, v16h b, v8f c) {
  return __builtin_amdgcn_wmma_f32_16x16x32_f16(false, a, false, b,
                                                (short)0, c, false, false);
}

// A-fragment (16x32 f16) from a row-major f16 row pointer (global or LDS).
// Lane L holds row M=L%16; halves h: K = base+h (h<8), base+16+(h-8) (h>=8),
// base = (L<16 ? 0 : 8).
__device__ __forceinline__ v16h load_a_f16(const _Float16* row, int k0, int lane) {
  int base = (lane < 16) ? 0 : 8;
  v8h x0 = *(const v8h*)(row + k0 + base);
  v8h x1 = *(const v8h*)(row + k0 + base + 16);
  v16h a;
#pragma unroll
  for (int i = 0; i < 8; ++i) { a[i] = x0[i]; a[8 + i] = x1[i]; }
  return a;
}

// B-fragment (32x16 f16): lane L holds column N=L%16; source is the
// TRANSPOSED operand (row N holds contiguous K). K = base+h (h<8), base+16+(h-8) (h>=8), base = (L<16?0:8).
__device__ __forceinline__ v16h load_b_f16(const _Float16* rowN, int k0, int lane) {
  int kb = k0 + ((lane < 16) ? 0 : 8);
  v8h b0 = *(const v8h*)(rowN + kb);
  v8h b1 = *(const v8h*)(rowN + kb + 16);
  v16h b;
#pragma unroll
  for (int i = 0; i < 8; ++i) { b[i] = b0[i]; b[8 + i] = b1[i]; }
  return b;
}

// ---------------- Kernel 1a: transpose + f32->f16 weights -----------------
__global__ void tcvt_kernel(const float* __restrict__ src,
                            _Float16* __restrict__ dst, int R, int Ccols) {
  int idx = blockIdx.x * 256 + threadIdx.x;
  if (idx < R * Ccols) {
    int r = idx / Ccols, c = idx % Ccols;
    dst[(size_t)c * R + r] = (_Float16)src[idx];
  }
}

// ---------------- Kernel 1b: x f32 -> f16 (one-time convert) --------------
__global__ void cvtx_kernel(const float* __restrict__ x,
                            _Float16* __restrict__ xh) {
  int idx = blockIdx.x * 256 + threadIdx.x;   // one thread per 8 elements
  v4f f0 = ((const v4f*)x)[2 * idx];
  v4f f1 = ((const v4f*)x)[2 * idx + 1];
  v8h h;
#pragma unroll
  for (int i = 0; i < 4; ++i) { h[i] = (_Float16)f0[i]; h[4 + i] = (_Float16)f1[i]; }
  ((v8h*)xh)[idx] = h;
}

// ---------------- Kernel 2: projection GEMM (WMMA, f16 in) ----------------
// out = xh[16384,512] @ W  via WT (f16, [Nout][512]).
// Each wave: one 16-row M tile x 32-wide N pair (A-fragment reuse).
// mode 0: out f16 row-major [16384][Nout]   (Q, K)
// mode 1: out f16 transposed-per-batch VT [B][512][4096]   (V)
__global__ __launch_bounds__(128) void proj_kernel(const _Float16* __restrict__ xh,
                                                   const _Float16* __restrict__ WT,
                                                   _Float16* __restrict__ out,
                                                   int Nout, int mode) {
  int wid    = blockIdx.x * 4 + (threadIdx.x >> 5);
  int lane   = threadIdx.x & 31;
  int npairs = Nout >> 5;
  int mt = wid / npairs, np = wid % npairs;
  int m0 = mt << 4, n0 = np << 5;

  const _Float16* xrow  = xh + (size_t)(m0 + (lane & 15)) * CD;
  const _Float16* wrow0 = WT + (size_t)(n0 + (lane & 15)) * CD;
  const _Float16* wrow1 = wrow0 + (size_t)16 * CD;

  v8f acc0 = {}, acc1 = {};
  for (int k = 0; k < CD; k += 32) {
    v16h a = load_a_f16(xrow, k, lane);
    acc0 = wmma16(a, load_b_f16(wrow0, k, lane), acc0);
    acc1 = wmma16(a, load_b_f16(wrow1, k, lane), acc1);
  }

  int hi = lane >> 4;
  int N  = lane & 15;
  if (mode == 0) {
#pragma unroll
    for (int r = 0; r < 8; ++r) {
      out[(size_t)(m0 + r + 8 * hi) * Nout + n0 + N]      = (_Float16)acc0[r];
      out[(size_t)(m0 + r + 8 * hi) * Nout + n0 + 16 + N] = (_Float16)acc1[r];
    }
  } else {
    int b     = m0 >> 12;
    int ibase = m0 & (HWD - 1);
    v8h pk0, pk1;
#pragma unroll
    for (int r = 0; r < 8; ++r) { pk0[r] = (_Float16)acc0[r]; pk1[r] = (_Float16)acc1[r]; }
    _Float16* dst0 = out + ((size_t)(b * CD + n0 + N)) * HWD + ibase + 8 * hi;
    _Float16* dst1 = dst0 + (size_t)16 * HWD;
    *(v8h*)dst0 = pk0;       // 8 consecutive i's -> one b128 store
    *(v8h*)dst1 = pk1;
  }
}

// ---------------- Kernel 3: fused attention -------------------------------
// grid = B * HW/32 blocks; block = 256 threads (8 waves).
// LDS: 32 x 4096 f16 score strip (256 KB) + 32x64 f16 Q stage (4 KB)
//      -> one workgroup per WGP (320 KB budget).
__global__ __launch_bounds__(256) void attn_kernel(const _Float16* __restrict__ Qh,
                                                   const _Float16* __restrict__ Kh,
                                                   const _Float16* __restrict__ VT,
                                                   float* __restrict__ OT) {
  extern __shared__ _Float16 sP[];           // [32][4096] then [32][64] Q
  _Float16* sQ = sP + (size_t)32 * HWD;
  __shared__ float red[32][9];
  __shared__ float rowmax[32];
  __shared__ float rowinv[32];

  int tid  = threadIdx.x;
  int lane = tid & 31;
  int wv   = tid >> 5;                       // wave 0..7
  int blk  = blockIdx.x;
  int bIdx = blk >> 7;                       // batch (128 blocks per batch)
  int i0   = (blk & 127) << 5;               // query row block (32 rows)

  // -- Stage Q block (32x64 f16 = 4KB contiguous) into LDS via async copy --
  {
    const _Float16* qsrc = Qh + (size_t)(bIdx * HWD + i0) * CR + tid * 8;
    unsigned ldsoff = (unsigned)(uintptr_t)(sQ + tid * 8);
    asm volatile("global_load_async_to_lds_b128 %0, %1, off"
                 :: "v"(ldsoff), "v"(qsrc) : "memory");
    asm volatile("s_wait_asynccnt 0x0" ::: "memory");
  }
  __syncthreads();

  // -- Q A-fragments for both 16-row tiles (from LDS) ----------------------
  const _Float16* qr0 = sQ + (size_t)(lane & 15) * CR;
  const _Float16* qr1 = sQ + (size_t)(16 + (lane & 15)) * CR;
  v16h aq[2][2];
  aq[0][0] = load_a_f16(qr0, 0, lane);  aq[0][1] = load_a_f16(qr0, 32, lane);
  aq[1][0] = load_a_f16(qr1, 0, lane);  aq[1][1] = load_a_f16(qr1, 32, lane);

  // -- Phase 1: S = Q K^T, wave wv covers j in [512*wv, 512*wv+512) --------
  const _Float16* Kb = Kh + (size_t)bIdx * HWD * CR;
  int hi = lane >> 4;
  for (int t = 0; t < 32; ++t) {
    int j0 = (wv << 9) + (t << 4);
    const _Float16* krow = Kb + (size_t)(j0 + (lane & 15)) * CR;
    v16h b0 = load_b_f16(krow, 0, lane);
    v16h b1 = load_b_f16(krow, 32, lane);
    int col = j0 + (lane & 15);
#pragma unroll
    for (int rt = 0; rt < 2; ++rt) {
      v8f s = {};
      s = wmma16(aq[rt][0], b0, s);
      s = wmma16(aq[rt][1], b1, s);
#pragma unroll
      for (int r = 0; r < 8; ++r)
        sP[(size_t)(16 * rt + r + 8 * hi) * HWD + col] = (_Float16)s[r];
    }
  }
  __syncthreads();

  // -- Phase 2: exact two-pass softmax on the LDS strip --------------------
  int rr = tid >> 3;                         // row 0..31
  int ss = tid & 7;                          // 512-wide segment
  {
    const _Float16* p = sP + (size_t)rr * HWD + ss * 512;
    float m = -3.0e38f;
    for (int k = 0; k < 512; ++k) m = fmaxf(m, (float)p[k]);
    red[rr][ss] = m;
  }
  __syncthreads();
  if (ss == 0) {
    float m = red[rr][0];
    for (int k = 1; k < 8; ++k) m = fmaxf(m, red[rr][k]);
    rowmax[rr] = m;
  }
  __syncthreads();
  {
    float m = rowmax[rr];
    _Float16* p = sP + (size_t)rr * HWD + ss * 512;
    float sum = 0.f;
    for (int k = 0; k < 512; ++k) {
      float e = __expf((float)p[k] - m);     // v_exp_f32
      p[k] = (_Float16)e;                    // overwrite S with P
      sum += e;
    }
    red[rr][ss] = sum;
  }
  __syncthreads();
  if (ss == 0) {
    float sm = 0.f;
    for (int k = 0; k < 8; ++k) sm += red[rr][k];
    rowinv[rr] = 1.0f / sm;
  }
  __syncthreads();

  // -- Phase 3: O = P V, wave wv owns channel chunk [64*wv, 64*wv+64) ------
  int c0 = wv << 6;
  v8f acc[2][4] = {};
  const _Float16* Vb    = VT + (size_t)bIdx * CD * HWD;
  const _Float16* prow0 = sP + (size_t)(lane & 15) * HWD;
  const _Float16* prow1 = sP + (size_t)(16 + (lane & 15)) * HWD;
  for (int j = 0; j < HWD; j += 32) {
    v16h ap0 = load_a_f16(prow0, j, lane);   // P from LDS (ds_load_b128)
    v16h ap1 = load_a_f16(prow1, j, lane);
#pragma unroll
    for (int n = 0; n < 4; ++n) {
      const _Float16* vrow = Vb + (size_t)(c0 + n * 16 + (lane & 15)) * HWD;
      v16h bp = load_b_f16(vrow, j, lane);
      if (j + 32 < HWD)
        __builtin_prefetch(vrow + j + 64, 0, 3);   // global_prefetch_b8
      acc[0][n] = wmma16(ap0, bp, acc[0][n]);
      acc[1][n] = wmma16(ap1, bp, acc[1][n]);
    }
  }

  // -- Epilogue: scale by 1/rowsum, store as O^T [C][HW] (== bcd layout) ---
#pragma unroll
  for (int rt = 0; rt < 2; ++rt) {
#pragma unroll
    for (int n = 0; n < 4; ++n) {
      int c = c0 + n * 16 + (lane & 15);
      float* orow = OT + (size_t)(bIdx * CD + c) * HWD + i0 + 16 * rt + 8 * hi;
      v4f o0, o1;
#pragma unroll
      for (int r = 0; r < 4; ++r) o0[r] = acc[rt][n][r]     * rowinv[16 * rt + r + 8 * hi];
#pragma unroll
      for (int r = 0; r < 4; ++r) o1[r] = acc[rt][n][4 + r] * rowinv[16 * rt + 4 + r + 8 * hi];
      *(v4f*)orow       = o0;
      *((v4f*)orow + 1) = o1;
    }
  }
}

// ---------------- Kernel 4: epilogue out = gamma*OT + x -------------------
__global__ void epilogue_kernel(const float* __restrict__ OT,
                                const float* __restrict__ x,
                                const float* __restrict__ gamma,
                                float* __restrict__ out, int total4) {
  int idx = blockIdx.x * 256 + threadIdx.x;
  if (idx < total4) {
    float g = gamma[0];
    v4f o  = ((const v4f*)OT)[idx];
    v4f xi = ((const v4f*)x)[idx];
    v4f r;
#pragma unroll
    for (int i = 0; i < 4; ++i) r[i] = g * o[i] + xi[i];
    ((v4f*)out)[idx] = r;
  }
}

// ---------------- launch ---------------------------------------------------
extern "C" void kernel_launch(void* const* d_in, const int* in_sizes, int n_in,
                              void* d_out, int out_size, void* d_ws, size_t ws_size,
                              hipStream_t stream) {
  const float* x     = (const float*)d_in[0];
  const float* W1    = (const float*)d_in[1];
  const float* W2    = (const float*)d_in[2];
  const float* W3    = (const float*)d_in[3];
  const float* gamma = (const float*)d_in[4];
  float* out = (float*)d_out;
  char*  ws  = (char*)d_ws;

  // workspace layout (bytes)
  const size_t OFF_W1T = 0;                                   //  64x512 f16
  const size_t OFF_W2T = OFF_W1T + 64 * 512 * 2;              //  64x512 f16
  const size_t OFF_W3T = OFF_W2T + 64 * 512 * 2;              // 512x512 f16
  const size_t OFF_XH  = OFF_W3T + 512 * 512 * 2;             // 16384x512 f16
  const size_t OFF_Q   = OFF_XH + (size_t)16384 * 512 * 2;    // 16384x64 f16
  const size_t OFF_K   = OFF_Q  + (size_t)16384 * 64 * 2;     // 16384x64 f16
  const size_t OFF_VT  = OFF_K  + (size_t)16384 * 64 * 2;     // B x 512 x 4096 f16
  const size_t OFF_OT  = OFF_VT + (size_t)NB * CD * HWD * 2;  // B x 512 x 4096 f32

  _Float16* w1T = (_Float16*)(ws + OFF_W1T);
  _Float16* w2T = (_Float16*)(ws + OFF_W2T);
  _Float16* w3T = (_Float16*)(ws + OFF_W3T);
  _Float16* xh  = (_Float16*)(ws + OFF_XH);
  _Float16* Qh  = (_Float16*)(ws + OFF_Q);
  _Float16* Kh  = (_Float16*)(ws + OFF_K);
  _Float16* VT  = (_Float16*)(ws + OFF_VT);
  float*    OT  = (float*)   (ws + OFF_OT);

  // 1) weight transpose + convert, x convert (once)
  tcvt_kernel<<<(512 * 64 + 255) / 256, 256, 0, stream>>>(W1, w1T, 512, 64);
  tcvt_kernel<<<(512 * 64 + 255) / 256, 256, 0, stream>>>(W2, w2T, 512, 64);
  tcvt_kernel<<<(512 * 512 + 255) / 256, 256, 0, stream>>>(W3, w3T, 512, 512);
  cvtx_kernel<<<4096, 256, 0, stream>>>(x, xh);   // 8M elems / 8 per thread

  // 2) projections (pure-f16 WMMA): 1024 M-tiles x 32-wide N pairs
  proj_kernel<<<512,  128, 0, stream>>>(xh, w1T, Qh, 64, 0);
  proj_kernel<<<512,  128, 0, stream>>>(xh, w2T, Kh, 64, 0);
  proj_kernel<<<4096, 128, 0, stream>>>(xh, w3T, VT, 512, 1);

  // 3) fused attention: 4 batches x 128 query blocks of 32 rows
  size_t shmem = (size_t)32 * HWD * 2 + (size_t)32 * CR * 2;  // 260 KB
  attn_kernel<<<NB * (HWD / 32), 256, shmem, stream>>>(Qh, Kh, VT, OT);

  // 4) epilogue: out = gamma * bcd + x  (8M elements, float4)
  int total4 = NB * HWD * CD / 4;
  epilogue_kernel<<<(total4 + 255) / 256, 256, 0, stream>>>(OT, x, gamma, out, total4);
}
